// ScalarGraphGRU_5901285065213
// MI455X (gfx1250) — compile-verified, matrix operand order checked by reading, not run on hardware
//
#include <hip/hip_runtime.h>
#include <hip/hip_bf16.h>
#include <stdint.h>

typedef unsigned int  u32;
typedef unsigned short u16;
typedef __attribute__((ext_vector_type(16))) u16    u16x16;
typedef __attribute__((ext_vector_type(8)))  u16    u16x8;
typedef __attribute__((ext_vector_type(16))) __bf16 bf16x16;
typedef __attribute__((ext_vector_type(8)))  float  v8f;

#define NG   2048          // n_graphs
#define VV   16            // max_vertices
#define HH   256           // hidden
#define G3H  768           // 3*H
#define LHID 128           // MLP hidden
#define AA   8             // actions
#define MT   16            // graphs per block (one WMMA M-tile)
#define NBLK (NG / MT)     // 128 blocks

// ---------------- bf16 helpers (RNE) ----------------
__device__ __forceinline__ u16 f2bf(float f) {
    u32 u = __builtin_bit_cast(u32, f);
    u32 r = (u + 0x7FFFu + ((u >> 16) & 1u)) >> 16;
    return (u16)r;
}
__device__ __forceinline__ float bf2f(u16 h) {
    u32 u = ((u32)h) << 16;
    return __builtin_bit_cast(float, u);
}

// ---------------- threefry2x32 (JAX-compatible) ----------------
__device__ __forceinline__ u32 rotl32(u32 x, int r) { return (x << r) | (x >> (32 - r)); }
__device__ __forceinline__ void tf_rounds(u32& x0, u32& x1, int r0, int r1, int r2, int r3) {
    x0 += x1; x1 = rotl32(x1, r0); x1 ^= x0;
    x0 += x1; x1 = rotl32(x1, r1); x1 ^= x0;
    x0 += x1; x1 = rotl32(x1, r2); x1 ^= x0;
    x0 += x1; x1 = rotl32(x1, r3); x1 ^= x0;
}
__device__ __forceinline__ void tf2x32(u32 k0, u32 k1, u32 x0, u32 x1, u32& o0, u32& o1) {
    u32 k2 = k0 ^ k1 ^ 0x1BD11BDAu;
    x0 += k0; x1 += k1;
    tf_rounds(x0, x1, 13, 15, 26, 6);  x0 += k1; x1 += k2 + 1u;
    tf_rounds(x0, x1, 17, 29, 16, 24); x0 += k2; x1 += k0 + 2u;
    tf_rounds(x0, x1, 13, 15, 26, 6);  x0 += k0; x1 += k1 + 3u;
    tf_rounds(x0, x1, 17, 29, 16, 24); x0 += k1; x1 += k2 + 4u;
    tf_rounds(x0, x1, 13, 15, 26, 6);  x0 += k2; x1 += k0 + 5u;
    o0 = x0; o1 = x1;
}
__device__ __forceinline__ void fold_in(u32 k0, u32 k1, u32 d, u32& n0, u32& n1) {
    tf2x32(k0, k1, 0u, d, n0, n1);
}
// JAX random_bits flat-index layout: pair (i, i+n/2)
__device__ __forceinline__ u32 rbits(u32 k0, u32 k1, u32 i, u32 n) {
    u32 h = n >> 1, o0, o1;
    if (i < h) { tf2x32(k0, k1, i, i + h, o0, o1); return o0; }
    tf2x32(k0, k1, i - h, i, o0, o1); return o1;
}
__device__ __forceinline__ float gumbel_from_bits(u32 b) {
    float f = __builtin_bit_cast(float, 0x3F800000u | (b >> 9)) - 1.0f; // [0,1)
    const float tiny = 1.1754944e-38f;
    f = f * (1.0f - tiny) + tiny;                                       // [tiny,1)
    return -logf(-logf(f));
}
// categorical + log_softmax[sample]
__device__ __forceinline__ int sample_cat(u32 k0, u32 k1, const float* lg, int C,
                                          u32 flat_base, u32 ntotal, float& lp) {
    int best = 0; float bestv = -3.4e38f;
    for (int c = 0; c < C; ++c) {
        u32 b = rbits(k0, k1, flat_base + (u32)c, ntotal);
        float v = lg[c] + gumbel_from_bits(b);
        if (v > bestv) { bestv = v; best = c; }
    }
    float m = lg[0];
    for (int c = 1; c < C; ++c) m = fmaxf(m, lg[c]);
    float s = 0.f;
    for (int c = 0; c < C; ++c) s += expf(lg[c] - m);
    lp = lg[best] - (m + logf(s));
    return best;
}

// ---------------- WMMA GEMM: out[16 x NT*16] = A(bf16,LDS) x Bfrags(bf16,global) ----------------
// A: [16][As] ushort (bf16), M=16 rows, K = KT*32.
// B: pre-swizzled fragments, frag = nt*KT + kt, each frag = 32 lanes * 16 bf16 (1 KB).
// Dual accumulator chains per wave iteration for WMMA ILP (A fragments shared).
__device__ __forceinline__ void gemm16(const u16* A, int As, const u16* B, int KT, int NT,
                                       float* out, int Os, int tid) {
    const int w = tid >> 5, lane = tid & 31;
    const int m = lane & 15, half = lane >> 4;
    const u16* ap0 = A + m * As + half * 8;
    int nt = w;
    for (; nt + 8 < NT; nt += 16) {
        v8f acc0 = {}, acc1 = {};
        const u16* bp0 = B + (size_t)nt * KT * 512 + lane * 16;
        const u16* bp1 = B + (size_t)(nt + 8) * KT * 512 + lane * 16;
        for (int kt = 0; kt < KT; ++kt) {
            u16x8 alo = *(const u16x8*)(ap0 + kt * 32);
            u16x8 ahi = *(const u16x8*)(ap0 + kt * 32 + 16);
            u16x16 av;
#pragma unroll
            for (int i = 0; i < 8; ++i) { av[i] = alo[i]; av[i + 8] = ahi[i]; }
            u16x16 bv0 = *(const u16x16*)(bp0 + kt * 512);
            u16x16 bv1 = *(const u16x16*)(bp1 + kt * 512);
            bf16x16 a = __builtin_bit_cast(bf16x16, av);
            acc0 = __builtin_amdgcn_wmma_f32_16x16x32_bf16(
                false, a, false, __builtin_bit_cast(bf16x16, bv0), (short)0, acc0, false, false);
            acc1 = __builtin_amdgcn_wmma_f32_16x16x32_bf16(
                false, a, false, __builtin_bit_cast(bf16x16, bv1), (short)0, acc1, false, false);
        }
        {
            float* op0 = out + (size_t)(half * 8) * Os + nt * 16 + m;
            float* op1 = out + (size_t)(half * 8) * Os + (nt + 8) * 16 + m;
#pragma unroll
            for (int r = 0; r < 8; ++r) {
                op0[(size_t)r * Os] = acc0[r];
                op1[(size_t)r * Os] = acc1[r];
            }
        }
    }
    for (; nt < NT; nt += 8) {
        v8f acc = {};
        const u16* bp = B + (size_t)nt * KT * 512 + lane * 16;
        for (int kt = 0; kt < KT; ++kt) {
            u16x8 alo = *(const u16x8*)(ap0 + kt * 32);
            u16x8 ahi = *(const u16x8*)(ap0 + kt * 32 + 16);
            u16x16 av;
#pragma unroll
            for (int i = 0; i < 8; ++i) { av[i] = alo[i]; av[i + 8] = ahi[i]; }
            u16x16 bv = *(const u16x16*)(bp + kt * 512);
            acc = __builtin_amdgcn_wmma_f32_16x16x32_bf16(
                false, __builtin_bit_cast(bf16x16, av),
                false, __builtin_bit_cast(bf16x16, bv),
                (short)0, acc, false, false);
        }
        float* op = out + (size_t)(half * 8) * Os + nt * 16 + m;
#pragma unroll
        for (int r = 0; r < 8; ++r) op[(size_t)r * Os] = acc[r];
    }
}

// ---------------- GRU elementwise ----------------
// scalar-x GRU: gi = x * wih + bih ; h_new -> dst (bf16)
__device__ __forceinline__ void gru_elem_sx(u16* dst, int ds, const u16* src, int ss,
                                            const float* gh, const float* ein,
                                            const float* wih, const float* bih,
                                            const float* bhh, int tid) {
    for (int it = tid; it < MT * HH; it += 256) {
        int g = it >> 8, j = it & 255;
        float x = ein[g];
        const float* gb = gh + g * G3H;
        float hr = gb[j]       + bhh[j];
        float hz = gb[256 + j] + bhh[256 + j];
        float hn = gb[512 + j] + bhh[512 + j];
        float ir = x * wih[j]       + bih[j];
        float iz = x * wih[256 + j] + bih[256 + j];
        float in_ = x * wih[512 + j] + bih[512 + j];
        float r = 1.f / (1.f + expf(-(ir + hr)));
        float z = 1.f / (1.f + expf(-(iz + hz)));
        float n = tanhf(in_ + r * hn);
        float ho = bf2f(src[g * ss + j]);
        dst[g * ds + j] = f2bf((1.f - z) * n + z * ho);
    }
}
// vector-x GRU: gi from gx buffer (x @ wih.T), h updated in place
__device__ __forceinline__ void gru_elem_vx(u16* h, int hs, const float* gh, const float* gx,
                                            const float* bih, const float* bhh, int tid) {
    for (int it = tid; it < MT * HH; it += 256) {
        int g = it >> 8, j = it & 255;
        const float* ghg = gh + g * G3H;
        const float* gxg = gx + g * G3H;
        float hr = ghg[j]       + bhh[j];
        float hz = ghg[256 + j] + bhh[256 + j];
        float hn = ghg[512 + j] + bhh[512 + j];
        float ir = gxg[j]       + bih[j];
        float iz = gxg[256 + j] + bih[256 + j];
        float in_ = gxg[512 + j] + bih[512 + j];
        float r = 1.f / (1.f + expf(-(ir + hr)));
        float z = 1.f / (1.f + expf(-(iz + hz)));
        float n = tanhf(in_ + r * hn);
        float ho = bf2f(h[g * hs + j]);
        h[g * hs + j] = f2bf((1.f - z) * n + z * ho);
    }
}

// ---------------- MLP layer 2: logits from raw hidden (bias+relu fused) ----------------
__device__ __forceinline__ void mlp2(const float* hid, const float* b1, const float* w2,
                                     const float* b2, int C, float* logits, int tid) {
    if (tid < MT * C) {
        int g = tid / C, c = tid % C;
        const float* hg = hid + g * LHID;
        const float* wc = w2 + c * LHID;
        float s = b2[c];
        for (int k = 0; k < LHID; ++k) {
            float hv = hg[k] + b1[k];
            s += fmaxf(hv, 0.f) * wc[k];
        }
        logits[g * AA + c] = s;
    }
}

// ---------------- weight pre-swizzle: fp32 (N x Krow) -> bf16 WMMA B fragments ----------------
__global__ __launch_bounds__(256) void pack_b(const float* __restrict__ W, u16* __restrict__ out,
                                              int Nrows, int Krow, int Kreal, int KT, int NT) {
    int t = blockIdx.x * blockDim.x + threadIdx.x;
    int total = NT * KT * 32;
    if (t >= total) return;
    int lane = t & 31, frag = t >> 5;
    int kt = frag % KT, nt = frag / KT;
    int half = lane >> 4, n = nt * 16 + (lane & 15);
    u16* o = out + (size_t)frag * 512 + lane * 16;
#pragma unroll
    for (int i = 0; i < 16; ++i) {
        int k = kt * 32 + half * 8 + (i < 8 ? i : i + 8);  // {0..7, 16..23} within 32-chunk
        float f = (n < Nrows && k < Kreal) ? W[(size_t)n * Krow + k] : 0.0f;
        o[i] = f2bf(f);
    }
}

// ---------------- main persistent kernel: 16 graphs / block ----------------
struct Params {
    const float *v_wih, *v_whh, *v_bih, *v_bhh;
    const float *e_wih, *e_whh, *e_bih, *e_bhh;
    const float *a_wih, *a_whh, *a_bih, *a_bhh;
    const float *vl_w1, *vl_b1, *vl_w2, *vl_b2;
    const float *el_w1, *el_b1, *el_w2, *el_b2;
    const float *al_w1, *al_b1, *al_w2, *al_b2;
    const float *hinit, *vinit;
    const u16 *p_vwhh, *p_ewhh, *p_awhh, *p_vwih, *p_vlw1, *p_elw1, *p_alw1;
    float *o_nextact, *o_conn, *o_activ, *o_lp;
};

__global__ __launch_bounds__(256) void graphgru_main(Params P) {
    __shared__ u16   vh[MT * 272];        // vertex hidden, bf16, stride 272
    __shared__ u16   eh[MT * 272];        // edge hidden
    __shared__ u16   vin[MT * 304];       // vertex input (K padded to 288); reused as 'ah'
    __shared__ float gbuf[MT * G3H];      // h @ whh.T
    __shared__ float gxbuf[MT * G3H];     // x @ wih.T ; reused as MLP hidden (stride 128)
    __shared__ float logitsb[MT * AA];
    __shared__ float einb[MT];
    __shared__ float connb[MT * 16];
    __shared__ float lp_v[MT];
    __shared__ float lp_tot[MT];
    __shared__ int   gatev[MT];
    __shared__ int   activeb[MT];
    __shared__ int   actb[MT];

    const int tid = threadIdx.x;
    const int G0  = blockIdx.x * MT;

    // init state
    for (int it = tid; it < MT * 288; it += 256) {
        int g = it / 288, k = it % 288;
        vin[g * 304 + k] = f2bf(k < 257 ? P.vinit[k] : 0.f);
    }
    for (int it = tid; it < MT * HH; it += 256) {
        int g = it >> 8, k = it & 255;
        vh[g * 272 + k] = f2bf(P.hinit[k]);
    }
    if (tid < MT) { lp_tot[tid] = 0.f; activeb[tid] = 1; }
    __syncthreads();

    for (int v = 0; v < VV; ++v) {
        if (tid < MT) { gatev[tid] = activeb[tid]; lp_v[tid] = 0.f; }
        __syncthreads();

        // ---- vertex GRU: vh = GRU(vin, vh) ----
        gemm16(vh, 272, P.p_vwhh, 8, 48, gbuf, G3H, tid);
        gemm16(vin, 304, P.p_vwih, 9, 48, gxbuf, G3H, tid);
        __syncthreads();
        gru_elem_vx(vh, 272, gbuf, gxbuf, P.v_bih, P.v_bhh, tid);
        __syncthreads();

        // ---- sample add_vertex ----
        gemm16(vh, 272, P.p_vlw1, 8, 8, gxbuf, LHID, tid);
        __syncthreads();
        mlp2(gxbuf, P.vl_b1, P.vl_w2, P.vl_b2, 2, logitsb, tid);
        __syncthreads();
        if (tid < MT) {
            u32 kv0, kv1, ks0, ks1;
            fold_in(0u, 42u, (u32)v, kv0, kv1);
            fold_in(kv0, kv1, 0u, ks0, ks1);
            float lp;
            int s = sample_cat(ks0, ks1, &logitsb[tid * AA], 2,
                               (u32)(G0 + tid) * 2u, (u32)NG * 2u, lp);
            lp_v[tid] += lp;
            activeb[tid] = activeb[tid] & (s > 0 ? 1 : 0);
        }
        // eh = vh, ein = 0
        for (int it = tid; it < MT * HH; it += 256) {
            int g = it >> 8, k = it & 255;
            eh[g * 272 + k] = vh[g * 272 + k];
        }
        if (tid < MT) einb[tid] = 0.f;
        __syncthreads();

        // ---- edge loop (mask e<v is uniform: only run live steps) ----
        for (int e = 0; e < v; ++e) {
            gemm16(eh, 272, P.p_ewhh, 8, 48, gbuf, G3H, tid);
            __syncthreads();
            gru_elem_sx(eh, 272, eh, 272, gbuf, einb, P.e_wih, P.e_bih, P.e_bhh, tid);
            __syncthreads();
            gemm16(eh, 272, P.p_elw1, 8, 8, gxbuf, LHID, tid);
            __syncthreads();
            mlp2(gxbuf, P.el_b1, P.el_w2, P.el_b2, 2, logitsb, tid);
            __syncthreads();
            if (tid < MT) {
                u32 kv0, kv1, ks0, ks1;
                fold_in(0u, 42u, (u32)v, kv0, kv1);
                fold_in(kv0, kv1, (u32)(e + 1), ks0, ks1);
                float lp;
                int c = sample_cat(ks0, ks1, &logitsb[tid * AA], 2,
                                   (u32)(G0 + tid) * 2u, (u32)NG * 2u, lp);
                lp_v[tid] += lp;
                einb[tid] = (float)c;
                connb[tid * 16 + e] = (float)c;
            }
            __syncthreads();
        }

        // ---- append GRU: ah = GRU(ein, eh)  (ah lives in vin region, stride 304) ----
        gemm16(eh, 272, P.p_awhh, 8, 48, gbuf, G3H, tid);
        __syncthreads();
        gru_elem_sx(vin, 304, eh, 272, gbuf, einb, P.a_wih, P.a_bih, P.a_bhh, tid);
        __syncthreads();

        // ---- sample activation (8-way) ----
        gemm16(vin, 304, P.p_alw1, 8, 8, gxbuf, LHID, tid);
        __syncthreads();
        mlp2(gxbuf, P.al_b1, P.al_w2, P.al_b2, 8, logitsb, tid);
        __syncthreads();
        if (tid < MT) {
            u32 kv0, kv1, ks0, ks1;
            fold_in(0u, 42u, (u32)v, kv0, kv1);
            fold_in(kv0, kv1, (u32)VV, ks0, ks1);
            float lp;
            int a = sample_cat(ks0, ks1, &logitsb[tid * AA], 8,
                               (u32)(G0 + tid) * 8u, (u32)NG * 8u, lp);
            lp_v[tid] += lp;
            actb[tid] = a;
        }
        __syncthreads();

        // ---- vin = concat(eh, act, zero-pad) ----
        for (int it = tid; it < MT * 288; it += 256) {
            int g = it / 288, k = it % 288;
            u16 val;
            if (k < 256)       val = eh[g * 272 + k];
            else if (k == 256) val = f2bf((float)actb[g]);
            else               val = 0;
            vin[g * 304 + k] = val;
        }

        // ---- outputs for this vertex ----
        if (tid < MT) {
            int G = G0 + tid;
            P.o_nextact[(size_t)G * VV + v] = activeb[tid] ? 1.f : 0.f;
            P.o_activ[(size_t)G * VV + v]   = (float)actb[tid];
            if (gatev[tid]) lp_tot[tid] += lp_v[tid];
        }
        {
            int g = tid >> 4, e = tid & 15;  // 256 threads = MT*16 exactly
            float cv = (e < v) ? connb[g * 16 + e] : 0.f;
            P.o_conn[((size_t)(G0 + g) * VV + v) * VV + e] = cv;
        }
        __syncthreads();
    }
    if (tid < MT) P.o_lp[G0 + tid] = lp_tot[tid];
}

// ---------------- host launch ----------------
extern "C" void kernel_launch(void* const* d_in, const int* in_sizes, int n_in,
                              void* d_out, int out_size, void* d_ws, size_t ws_size,
                              hipStream_t stream) {
    const float* v_wih = (const float*)d_in[0];
    const float* v_whh = (const float*)d_in[1];
    const float* v_bih = (const float*)d_in[2];
    const float* v_bhh = (const float*)d_in[3];
    const float* e_wih = (const float*)d_in[4];
    const float* e_whh = (const float*)d_in[5];
    const float* e_bih = (const float*)d_in[6];
    const float* e_bhh = (const float*)d_in[7];
    const float* a_wih = (const float*)d_in[8];
    const float* a_whh = (const float*)d_in[9];
    const float* a_bih = (const float*)d_in[10];
    const float* a_bhh = (const float*)d_in[11];
    const float* vl_w1 = (const float*)d_in[12];
    const float* vl_b1 = (const float*)d_in[13];
    const float* vl_w2 = (const float*)d_in[14];
    const float* vl_b2 = (const float*)d_in[15];
    const float* el_w1 = (const float*)d_in[16];
    const float* el_b1 = (const float*)d_in[17];
    const float* el_w2 = (const float*)d_in[18];
    const float* el_b2 = (const float*)d_in[19];
    const float* al_w1 = (const float*)d_in[20];
    const float* al_b1 = (const float*)d_in[21];
    const float* al_w2 = (const float*)d_in[22];
    const float* al_b2 = (const float*)d_in[23];
    const float* hinit = (const float*)d_in[24];
    const float* vinit = (const float*)d_in[25];

    // workspace layout (ushort units)
    u16* ws = (u16*)d_ws;
    u16* p_vwhh = ws + 0;        // 48*8*512 = 196608
    u16* p_ewhh = ws + 196608;
    u16* p_awhh = ws + 393216;
    u16* p_vwih = ws + 589824;   // 48*9*512 = 221184
    u16* p_vlw1 = ws + 811008;   // 8*8*512 = 32768
    u16* p_elw1 = ws + 843776;
    u16* p_alw1 = ws + 876544;   // end = 909312 ushorts (~1.78 MB)

    auto pack = [&](const float* W, u16* o, int Nr, int Krow, int Kreal, int KT, int NT) {
        int tot = NT * KT * 32;
        hipLaunchKernelGGL(pack_b, dim3((tot + 255) / 256), dim3(256), 0, stream,
                           W, o, Nr, Krow, Kreal, KT, NT);
    };
    pack(v_whh, p_vwhh, 768, 256, 256, 8, 48);
    pack(e_whh, p_ewhh, 768, 256, 256, 8, 48);
    pack(a_whh, p_awhh, 768, 256, 256, 8, 48);
    pack(v_wih, p_vwih, 768, 257, 257, 9, 48);   // K padded 257 -> 288 with zeros
    pack(vl_w1, p_vlw1, 128, 256, 256, 8, 8);
    pack(el_w1, p_elw1, 128, 256, 256, 8, 8);
    pack(al_w1, p_alw1, 128, 256, 256, 8, 8);

    Params P;
    P.v_wih = v_wih; P.v_whh = v_whh; P.v_bih = v_bih; P.v_bhh = v_bhh;
    P.e_wih = e_wih; P.e_whh = e_whh; P.e_bih = e_bih; P.e_bhh = e_bhh;
    P.a_wih = a_wih; P.a_whh = a_whh; P.a_bih = a_bih; P.a_bhh = a_bhh;
    P.vl_w1 = vl_w1; P.vl_b1 = vl_b1; P.vl_w2 = vl_w2; P.vl_b2 = vl_b2;
    P.el_w1 = el_w1; P.el_b1 = el_b1; P.el_w2 = el_w2; P.el_b2 = el_b2;
    P.al_w1 = al_w1; P.al_b1 = al_b1; P.al_w2 = al_w2; P.al_b2 = al_b2;
    P.hinit = hinit; P.vinit = vinit;
    P.p_vwhh = p_vwhh; P.p_ewhh = p_ewhh; P.p_awhh = p_awhh; P.p_vwih = p_vwih;
    P.p_vlw1 = p_vlw1; P.p_elw1 = p_elw1; P.p_alw1 = p_alw1;

    float* out = (float*)d_out;
    P.o_nextact = out;
    P.o_conn    = out + (size_t)NG * VV;
    P.o_activ   = out + (size_t)NG * VV + (size_t)NG * VV * VV;
    P.o_lp      = out + (size_t)NG * VV + (size_t)NG * VV * VV + (size_t)NG * VV;

    hipLaunchKernelGGL(graphgru_main, dim3(NBLK), dim3(256), 0, stream, P);
}
